// SwinTransformerBlock_45741401702827
// MI455X (gfx1250) — matrix operand order repaired and checked by reading, not compiled, not run on hardware
//
#include <hip/hip_runtime.h>
#include <math.h>

// ---------------------------------------------------------------------------
// Types for CDNA5 WMMA (gfx1250, wave32)
// ---------------------------------------------------------------------------
typedef __attribute__((ext_vector_type(16))) __bf16        v16bf;
typedef __attribute__((ext_vector_type(8)))  float         v8f;
typedef __attribute__((ext_vector_type(4)))  unsigned int  v4u;

union FragBF { v16bf v; v4u q[2]; };

// Async global->LDS (CDNA5 GLOBAL_LOAD_ASYNC_TO_LDS_B128, ASYNCcnt-tracked)
#if defined(__AMDGCN__) && __has_builtin(__builtin_amdgcn_global_load_async_to_lds_b128)
#define HAVE_ASYNC_LDS 1
typedef int v4i_t __attribute__((vector_size(4 * sizeof(int))));
typedef __attribute__((address_space(1))) v4i_t gv4i_t;   // global int4*
typedef __attribute__((address_space(3))) v4i_t lv4i_t;   // LDS int4*
#else
#define HAVE_ASYNC_LDS 0
#endif

#if HAVE_ASYNC_LDS
__device__ __forceinline__ void async_copy16(const void* gsrc, void* ldst) {
  __builtin_amdgcn_global_load_async_to_lds_b128(
      (gv4i_t*)(unsigned long long)(size_t)gsrc,
      (lv4i_t*)(unsigned int)(size_t)ldst, 0, 0);
}
__device__ __forceinline__ void wait_async_le(int n) {
#if __has_builtin(__builtin_amdgcn_s_wait_asynccnt)
  if (n == 0) __builtin_amdgcn_s_wait_asynccnt(0);
  else        __builtin_amdgcn_s_wait_asynccnt(1);
#else
  if (n == 0) asm volatile("s_wait_asynccnt 0x0" ::: "memory");
  else        asm volatile("s_wait_asynccnt 0x1" ::: "memory");
#endif
}
#endif

__device__ __forceinline__ unsigned short f2bf(float f) {
  union { float f; unsigned int u; } x; x.f = f;
  unsigned int u = x.u;
  if ((u & 0x7fffffffu) > 0x7f800000u) return (unsigned short)((u >> 16) | 0x40u); // NaN
  u += 0x7fffu + ((u >> 16) & 1u);   // round-to-nearest-even
  return (unsigned short)(u >> 16);
}
__device__ __forceinline__ float bf2f(unsigned short h) {
  union { unsigned int u; float f; } x; x.u = ((unsigned int)h) << 16;
  return x.f;
}

// ---------------------------------------------------------------------------
// Generic converter f32 -> bf16
// ---------------------------------------------------------------------------
__global__ void f32_to_bf16(const float* __restrict__ s, unsigned short* __restrict__ d, int n) {
  int e = blockIdx.x * 256 + threadIdx.x;
  if (e < n) d[e] = f2bf(s[e]);
}

// ---------------------------------------------------------------------------
// Setup: CPB table (225 x 16) and per-head logit scales
// ---------------------------------------------------------------------------
__device__ __forceinline__ float cpb_coord(int r) {
  float v = (float)(r - 7) * (8.0f / 7.0f);
  float t = log2f(fabsf(v) + 1.0f) * (1.0f / log2f(8.0f));
  return v < 0.0f ? -t : t;
}

__global__ void setup_tbl(const float* __restrict__ w1, const float* __restrict__ w2,
                          const float* __restrict__ ls,
                          float* __restrict__ tbl, float* __restrict__ scales) {
  int e = blockIdx.x * 256 + threadIdx.x;
  if (e < 3600) {
    int pos = e >> 4, h = e & 15;
    float t0 = cpb_coord(pos / 15), t1 = cpb_coord(pos % 15);
    float acc = 0.0f;
    for (int j = 0; j < 512; ++j) {
      float z  = t0 * w1[j * 2] + t1 * w1[j * 2 + 1];
      float hs = z * fminf(fmaxf(z + 3.0f, 0.0f), 6.0f) * (1.0f / 6.0f);  // hard_swish
      acc += w2[h * 512 + j] * hs;
    }
    tbl[pos * 16 + h] = acc;
  } else if (e < 3616) {
    int h = e - 3600;
    scales[h] = __expf(fminf(ls[h], logf(100.0f)));
  }
}

// ---------------------------------------------------------------------------
// LPU: xl = x + depthwise3x3(x)
// ---------------------------------------------------------------------------
__global__ void lpu_kernel(const float* __restrict__ x, const float* __restrict__ w,
                           float* __restrict__ xl) {
  const size_t e = (size_t)blockIdx.x * 256 + threadIdx.x;   // B*512*4096 elems
  const int hw = (int)(e & 4095);
  const int c  = (int)((e >> 12) & 511);
  const int b  = (int)(e >> 21);
  const int hh = hw >> 6, ww = hw & 63;
  const float* xb = x + ((size_t)b * 512 + c) * 4096;
  const float* wc = w + c * 9;
  float s = 0.0f;
  #pragma unroll
  for (int dy = -1; dy <= 1; ++dy)
    #pragma unroll
    for (int dx = -1; dx <= 1; ++dx) {
      int yy = hh + dy, xx = ww + dx;
      if (yy >= 0 && yy < 64 && xx >= 0 && xx < 64)
        s += xb[yy * 64 + xx] * wc[(dy + 1) * 3 + (dx + 1)];
    }
  xl[e] = x[e] + s;
}

// ---------------------------------------------------------------------------
// Roll(-4,-4 over tokens & channels) + window partition -> bf16 (65536 x 512)
// ---------------------------------------------------------------------------
__global__ void win_gather(const float* __restrict__ xl, unsigned short* __restrict__ xw) {
  const size_t e = (size_t)blockIdx.x * 256 + threadIdx.x;   // T*512 + c
  const int c = (int)(e & 511);
  const int T = (int)(e >> 9);
  const int win = T >> 6, n = T & 63;
  const int b = win >> 6, wb = win & 63;
  const int hh = ((wb >> 3) << 3) | (n >> 3);
  const int ww = ((wb & 7) << 3) | (n & 7);
  const int shw = (hh * 64 + ww + 4) & 4095;   // token roll -s
  const int sc  = (c + 4) & 511;               // channel roll -s (faithful)
  xw[e] = f2bf(xl[((size_t)b * 512 + sc) * 4096 + shw]);
}

// ---------------------------------------------------------------------------
// GEMM: D = A(bf16 MxK) @ Wt(bf16 stored NxK)^T, fp32 accumulate via WMMA.
// Workgroup = 8 waves, macro-tile 64(M) x 256(N).
// A tile double-buffered in LDS via GLOBAL_LOAD_ASYNC_TO_LDS_B128 (ASYNCcnt),
// overlapping the async fetch of tile k+1 with the 8 WMMAs of tile k.
// mode 0: fp32 token-major; mode 1: fp32 NCHW scatter (Cout); mode 2: bf16.
// ---------------------------------------------------------------------------
__global__ __launch_bounds__(256) void gemm_bf16_wmma(
    const unsigned short* __restrict__ A,
    const unsigned short* __restrict__ Wt,
    void* __restrict__ D,
    int M, int N, int K, int mode, int Cout) {
  __shared__ v4u AsQ[2][256];              // 2 x (64 rows x 32 cols bf16) = 8KB
  const int tid  = threadIdx.x;
  const int wave = tid >> 5;
  const int lane = tid & 31;
  const int ln   = lane & 15;
  const int hi   = lane >> 4;
  const int m0   = blockIdx.x * 64;
  const int n0   = blockIdx.y * 256 + wave * 32;
  (void)M;

  v8f zero = {};
  v8f acc[4][2];
  #pragma unroll
  for (int i = 0; i < 4; ++i) { acc[i][0] = zero; acc[i][1] = zero; }

  const unsigned short* aptr = A  + (size_t)(m0 + (tid >> 2)) * K + (tid & 3) * 8;
  const unsigned short* b0p  = Wt + (size_t)(n0 + ln) * K + hi * 16;
  const unsigned short* b1p  = Wt + (size_t)(n0 + 16 + ln) * K + hi * 16;

#if HAVE_ASYNC_LDS
  async_copy16(aptr, &AsQ[0][tid]);        // prologue: tile 0 -> buffer 0
#endif

  int cur = 0;
  for (int k0 = 0; k0 < K; k0 += 32) {
    const bool has_next = (k0 + 32) < K;
#if HAVE_ASYNC_LDS
    if (has_next) async_copy16(aptr + k0 + 32, &AsQ[cur ^ 1][tid]);  // prefetch k+1
    wait_async_le(has_next ? 1 : 0);       // tile k landed (in-order ASYNCcnt)
    __syncthreads();
#else
    __syncthreads();
    AsQ[cur][tid] = *reinterpret_cast<const v4u*>(aptr + k0);        // coop 16B/thread
    if (has_next) __builtin_prefetch(aptr + k0 + 32, 0, 1);          // global_prefetch_b8
    __syncthreads();
#endif

    FragBF bf0, bf1;   // B frag: lane=n col, hi selects K half (k = 16*hi + s here)
    bf0.q[0] = *reinterpret_cast<const v4u*>(b0p + k0);
    bf0.q[1] = *reinterpret_cast<const v4u*>(b0p + k0 + 8);
    bf1.q[0] = *reinterpret_cast<const v4u*>(b1p + k0);
    bf1.q[1] = *reinterpret_cast<const v4u*>(b1p + k0 + 8);

    #pragma unroll
    for (int mi = 0; mi < 4; ++mi) {
      FragBF af;       // A frag: lane=m row, chunks at hi*16 and hi*16+8
      const int row = mi * 16 + ln;
      af.q[0] = AsQ[cur][row * 4 + 2 * hi];
      af.q[1] = AsQ[cur][row * 4 + 2 * hi + 1];
      acc[mi][0] = __builtin_amdgcn_wmma_f32_16x16x32_bf16(
          false, af.v, false, bf0.v, (short)0, acc[mi][0], false, false);
      acc[mi][1] = __builtin_amdgcn_wmma_f32_16x16x32_bf16(
          false, af.v, false, bf1.v, (short)0, acc[mi][1], false, false);
    }

#if HAVE_ASYNC_LDS
    __syncthreads();   // all waves done reading buf[cur] before it is re-filled
#endif
    cur ^= 1;
  }

  #pragma unroll
  for (int mi = 0; mi < 4; ++mi)
    #pragma unroll
    for (int nj = 0; nj < 2; ++nj) {
      const int n = n0 + nj * 16 + ln;
      #pragma unroll
      for (int r = 0; r < 8; ++r) {
        const int m = m0 + mi * 16 + hi * 8 + r;   // D layout: M = r + 8*hi
        const float v = acc[mi][nj][r];
        if (mode == 0) {
          ((float*)D)[(size_t)m * N + n] = v;
        } else if (mode == 2) {
          ((unsigned short*)D)[(size_t)m * N + n] = f2bf(v);
        } else {
          const int b = m >> 12, hw = m & 4095;
          ((float*)D)[((size_t)b * Cout + n) * 4096 + hw] = v;
        }
      }
    }
}

// ---------------------------------------------------------------------------
// QKV rearrange + cosine norm. Block = 512 thr = 16 waves (one wave per head).
// qn,kn -> [win][head][tok][32] bf16 ; v -> transposed [win][head][32][tok].
// ---------------------------------------------------------------------------
__global__ __launch_bounds__(512) void qkv_rearrange(
    const unsigned short* __restrict__ qkv,
    unsigned short* __restrict__ qn, unsigned short* __restrict__ kn,
    unsigned short* __restrict__ vT) {
  const int T = blockIdx.x;                   // token 0..65535
  const int h = threadIdx.x >> 5, d = threadIdx.x & 31;
  const unsigned short* base = qkv + (size_t)T * 1536;
  const float q = bf2f(base[h * 32 + d]);
  const float k = bf2f(base[512 + h * 32 + d]);
  const float v = bf2f(base[1024 + h * 32 + d]);
  float qs = q * q, ks = k * k;
  #pragma unroll
  for (int off = 16; off; off >>= 1) { qs += __shfl_xor(qs, off); ks += __shfl_xor(ks, off); }
  const float qd = fmaxf(sqrtf(qs), 1e-12f);
  const float kd = fmaxf(sqrtf(ks), 1e-12f);
  const int win = T >> 6, n = T & 63;
  const size_t hb = (size_t)(win * 16 + h);
  qn[(hb * 64 + n) * 32 + d] = f2bf(q / qd);
  kn[(hb * 64 + n) * 32 + d] = f2bf(k / kd);
  vT[(hb * 32 + d) * 64 + n] = f2bf(v);
}

// ---------------------------------------------------------------------------
// Shift-window mask region (faithful 9-region construction, rows 56..59 -> 0)
// ---------------------------------------------------------------------------
__device__ __forceinline__ int mask_region(int r, int c) {
  int rr = (r < 56) ? 0 : ((r >= 60) ? 2 : -1);
  int cc = (c < 56) ? 0 : ((c >= 60) ? 2 : -1);
  return (rr < 0 || cc < 0) ? 0 : rr * 3 + cc;
}

// ---------------------------------------------------------------------------
// Attention logits: S = (qn @ kn^T)*scale + 16*sigmoid(cpb) + mask, bf16.
// One wave per 16x16 tile; one WMMA (K = hd = 32).
// ---------------------------------------------------------------------------
__global__ __launch_bounds__(32) void attn_logits(
    const unsigned short* __restrict__ qn, const unsigned short* __restrict__ kn,
    const float* __restrict__ tbl, const float* __restrict__ scales,
    unsigned short* __restrict__ S) {
  const int lane = threadIdx.x;
  const int tile = blockIdx.x & 15;
  const int head = blockIdx.x >> 4;
  const int win  = blockIdx.y;
  const int i0 = (tile >> 2) * 16, j0 = (tile & 3) * 16;
  const int ln = lane & 15, hi = lane >> 4;
  const unsigned short* qb = qn + (size_t)(win * 16 + head) * 2048;
  const unsigned short* kb = kn + (size_t)(win * 16 + head) * 2048;

  FragBF a, b;
  const unsigned short* ap = qb + (i0 + ln) * 32;
  a.q[0] = *reinterpret_cast<const v4u*>(ap + hi * 16);
  a.q[1] = *reinterpret_cast<const v4u*>(ap + hi * 16 + 8);
  const unsigned short* bp = kb + (j0 + ln) * 32 + hi * 16;
  b.q[0] = *reinterpret_cast<const v4u*>(bp);
  b.q[1] = *reinterpret_cast<const v4u*>(bp + 8);

  v8f c = {};
  c = __builtin_amdgcn_wmma_f32_16x16x32_bf16(false, a.v, false, b.v, (short)0, c, false, false);

  const float sc = scales[head];
  const int wb = win & 63, wih = wb >> 3, wjw = wb & 7;
  const int j = j0 + ln, jh = j >> 3, jw = j & 7;
  const int mj = mask_region(wih * 8 + jh, wjw * 8 + jw);
  #pragma unroll
  for (int r = 0; r < 8; ++r) {
    const int i = i0 + hi * 8 + r, ih = i >> 3, iw = i & 7;
    const int mi_ = mask_region(wih * 8 + ih, wjw * 8 + iw);
    const float t = tbl[((ih - jh + 7) * 15 + (iw - jw + 7)) * 16 + head];
    const float bias = 16.0f / (1.0f + __expf(-t));
    const float v = c[r] * sc + bias + ((mi_ != mj) ? -100.0f : 0.0f);
    S[(size_t)(win * 16 + head) * 4096 + i * 64 + j] = f2bf(v);
  }
}

// ---------------------------------------------------------------------------
// Faithful softmax over the HEAD axis (16 values, stride 4096), in-place bf16
// ---------------------------------------------------------------------------
__global__ void softmax_heads(unsigned short* __restrict__ S) {
  const size_t e = (size_t)blockIdx.x * 256 + threadIdx.x;   // win*4096 + ij
  const int win = (int)(e >> 12), ij = (int)(e & 4095);
  unsigned short* p = S + (size_t)win * 65536 + ij;
  float v[16], mx = -1e30f;
  #pragma unroll
  for (int h = 0; h < 16; ++h) { v[h] = bf2f(p[h * 4096]); mx = fmaxf(mx, v[h]); }
  float s = 0.0f;
  #pragma unroll
  for (int h = 0; h < 16; ++h) { v[h] = __expf(v[h] - mx); s += v[h]; }
  const float inv = 1.0f / s;
  #pragma unroll
  for (int h = 0; h < 16; ++h) p[h * 4096] = f2bf(v[h] * inv);
}

// ---------------------------------------------------------------------------
// PV: O = P(64x64) @ V(64x32) per (win,head); 2 WMMAs per 16x16 tile (K=64).
// Stores token-major fp32 at column head*32+d.
// ---------------------------------------------------------------------------
__global__ __launch_bounds__(32) void attn_pv(
    const unsigned short* __restrict__ P, const unsigned short* __restrict__ vT,
    float* __restrict__ O) {
  const int lane = threadIdx.x;
  const int tile = blockIdx.x & 7;
  const int head = blockIdx.x >> 3;
  const int win  = blockIdx.y;
  const int m0 = (tile >> 1) * 16, n0 = (tile & 1) * 16;
  const int ln = lane & 15, hi = lane >> 4;
  const unsigned short* Pb = P  + (size_t)(win * 16 + head) * 4096;  // 64x64
  const unsigned short* Vb = vT + (size_t)(win * 16 + head) * 2048;  // 32x64 (d-major)

  v8f c = {};
  #pragma unroll
  for (int k0 = 0; k0 < 64; k0 += 32) {
    FragBF a, b;
    const unsigned short* ap = Pb + (m0 + ln) * 64 + k0;
    a.q[0] = *reinterpret_cast<const v4u*>(ap + hi * 16);
    a.q[1] = *reinterpret_cast<const v4u*>(ap + hi * 16 + 8);
    const unsigned short* bp = Vb + (n0 + ln) * 64 + k0 + hi * 16;
    b.q[0] = *reinterpret_cast<const v4u*>(bp);
    b.q[1] = *reinterpret_cast<const v4u*>(bp + 8);
    c = __builtin_amdgcn_wmma_f32_16x16x32_bf16(false, a.v, false, b.v, (short)0, c, false, false);
  }
  #pragma unroll
  for (int r = 0; r < 8; ++r) {
    const int i = m0 + hi * 8 + r;
    O[((size_t)win * 64 + i) * 512 + head * 32 + n0 + ln] = c[r];
  }
}

// ---------------------------------------------------------------------------
// Block reduction helper (256 threads)
// ---------------------------------------------------------------------------
__device__ __forceinline__ float block_sum256(float v, float* red) {
  red[threadIdx.x] = v; __syncthreads();
  #pragma unroll
  for (int s = 128; s > 0; s >>= 1) {
    if (threadIdx.x < s) red[threadIdx.x] += red[threadIdx.x + s];
    __syncthreads();
  }
  float r = red[0]; __syncthreads();
  return r;
}

// ---------------------------------------------------------------------------
// Residual 1: x1 = rs1*x_temp + LN_c(un-partitioned o). Block = one pixel.
// Uses faithful buggy un-partition: p = (((wj*8+wi)*8+ti)*8+tj)*512 + c with
// roll(+4,+4) on (wi,wj); here inverted per output (c_out, hw).
// Writes x1 fp32 token-major and bf16 copy (GEMM A operand).
// ---------------------------------------------------------------------------
__global__ __launch_bounds__(256) void ln1_res(
    const float* __restrict__ ao, const float* __restrict__ xl,
    const float* __restrict__ g, const float* __restrict__ be,
    const float* __restrict__ rs1,
    float* __restrict__ x1, unsigned short* __restrict__ x1b) {
  __shared__ float red[256];
  const int pix = blockIdx.x;
  const int b = pix >> 12, hw = pix & 4095;
  const int tj = hw >> 9, c = hw & 511;
  float vals[2];
  #pragma unroll
  for (int t = 0; t < 2; ++t) {
    const int co = threadIdx.x + t * 256;
    const int wj = co >> 6, wi = (co >> 3) & 7, ti = co & 7;
    const int srow = (b * 64 + ((wi + 4) & 7) * 8 + ((wj + 4) & 7)) * 64 + ti * 8 + tj;
    vals[t] = ao[(size_t)srow * 512 + c];
  }
  const float mean = block_sum256(vals[0] + vals[1], red) * (1.0f / 512.0f);
  const float d0 = vals[0] - mean, d1 = vals[1] - mean;
  const float var = block_sum256(d0 * d0 + d1 * d1, red) * (1.0f / 512.0f);
  const float inv = rsqrtf(var + 8e-8f);
  #pragma unroll
  for (int t = 0; t < 2; ++t) {
    const int co = threadIdx.x + t * 256;
    const float lnv = (vals[t] - mean) * inv * g[co] + be[co];
    const float xt  = xl[((size_t)b * 512 + co) * 4096 + hw];
    const float o   = rs1[co] * xt + lnv;
    x1[(size_t)pix * 512 + co]  = o;
    x1b[(size_t)pix * 512 + co] = f2bf(o);
  }
}

// ---------------------------------------------------------------------------
// MLP: h2 = StarReLU(h + dwconv3x3(h)); h is NCHW fp32 (2048 ch), output bf16
// token-major for the pw2 GEMM.
// ---------------------------------------------------------------------------
__global__ void mlp_dw_star(const float* __restrict__ h, const float* __restrict__ dww,
                            const float* __restrict__ sr_s, const float* __restrict__ sr_b,
                            unsigned short* __restrict__ h2) {
  const size_t e = (size_t)blockIdx.x * 256 + threadIdx.x;   // B*2048*4096 elems
  const int hw = (int)(e & 4095);
  const int c  = (int)((e >> 12) & 2047);
  const int b  = (int)(e >> 23);
  const int hh = hw >> 6, ww = hw & 63;
  const float* hb = h + ((size_t)b * 2048 + c) * 4096;
  const float* wc = dww + c * 9;
  float s = 0.0f;
  #pragma unroll
  for (int dy = -1; dy <= 1; ++dy)
    #pragma unroll
    for (int dx = -1; dx <= 1; ++dx) {
      int yy = hh + dy, xx = ww + dx;
      if (yy >= 0 && yy < 64 && xx >= 0 && xx < 64)
        s += hb[yy * 64 + xx] * wc[(dy + 1) * 3 + (dx + 1)];
    }
  const float v0 = hb[hw] + s;
  const float r  = fmaxf(v0, 0.0f);
  const float v  = sr_s[0] * r * r + sr_b[0];
  h2[((size_t)b * 4096 + hw) * 2048 + c] = f2bf(v);
}

// ---------------------------------------------------------------------------
// Final: out = rs2*x1 + LN_c(y); y token-major fp32, out NCHW fp32.
// ---------------------------------------------------------------------------
__global__ __launch_bounds__(256) void ln2_out(
    const float* __restrict__ y, const float* __restrict__ x1,
    const float* __restrict__ g, const float* __restrict__ be,
    const float* __restrict__ rs2, float* __restrict__ out) {
  __shared__ float red[256];
  const int pix = blockIdx.x;
  const int b = pix >> 12, hw = pix & 4095;
  float vals[2];
  #pragma unroll
  for (int t = 0; t < 2; ++t)
    vals[t] = y[(size_t)pix * 512 + threadIdx.x + t * 256];
  const float mean = block_sum256(vals[0] + vals[1], red) * (1.0f / 512.0f);
  const float d0 = vals[0] - mean, d1 = vals[1] - mean;
  const float var = block_sum256(d0 * d0 + d1 * d1, red) * (1.0f / 512.0f);
  const float inv = rsqrtf(var + 8e-8f);
  #pragma unroll
  for (int t = 0; t < 2; ++t) {
    const int co = threadIdx.x + t * 256;
    const float lnv = (vals[t] - mean) * inv * g[co] + be[co];
    out[((size_t)b * 512 + co) * 4096 + hw] = rs2[co] * x1[(size_t)pix * 512 + co] + lnv;
  }
}

// ---------------------------------------------------------------------------
// Host launcher
// ---------------------------------------------------------------------------
extern "C" void kernel_launch(void* const* d_in, const int* in_sizes, int n_in,
                              void* d_out, int out_size, void* d_ws, size_t ws_size,
                              hipStream_t stream) {
  (void)in_sizes; (void)n_in; (void)out_size; (void)ws_size;
  const float* x      = (const float*)d_in[0];
  const float* lpu_w  = (const float*)d_in[1];
  const float* qkv_w  = (const float*)d_in[2];
  const float* out_w  = (const float*)d_in[3];
  const float* cpb_w1 = (const float*)d_in[4];
  const float* cpb_w2 = (const float*)d_in[5];
  const float* ls     = (const float*)d_in[6];
  const float* gamma1 = (const float*)d_in[7];
  const float* beta1  = (const float*)d_in[8];
  const float* gamma2 = (const float*)d_in[9];
  const float* beta2  = (const float*)d_in[10];
  const float* pw1_w  = (const float*)d_in[11];
  const float* dw_w   = (const float*)d_in[12];
  const float* pw2_w  = (const float*)d_in[13];
  const float* sr_s   = (const float*)d_in[14];
  const float* sr_b   = (const float*)d_in[15];
  const float* rs1    = (const float*)d_in[16];
  const float* rs2    = (const float*)d_in[17];

  char* ws = (char*)d_ws;
  size_t cur = 0;
  auto alloc = [&](size_t bytes) { size_t p = cur; cur += (bytes + 255) & ~(size_t)255; return p; };
  const size_t XL   = alloc(134217728ull);   // xl fp32 NCHW
  const size_t WQKV = alloc(1572864ull);     // weights bf16
  const size_t WOUT = alloc(524288ull);
  const size_t WPW1 = alloc(2097152ull);
  const size_t WPW2 = alloc(2097152ull);
  const size_t TBL  = alloc(14400ull);       // cpb table 225x16 fp32
  const size_t SCL  = alloc(64ull);          // per-head scales
  const size_t XWIN = alloc(67108864ull);    // bf16 tokens (x_win / o_bf16 / x1_bf16)
  const size_t QKV  = alloc(201326592ull);   // qkv bf16; later reused for h2 bf16
  const size_t SBUF = alloc(134217728ull);   // attn logits/probs bf16 (contiguous after QKV)
  const size_t QN   = alloc(67108864ull);
  const size_t KN   = alloc(67108864ull);
  const size_t VT   = alloc(67108864ull);
  const size_t OBUF = alloc(134217728ull);   // o f32 / attn_out f32 / y f32
  const size_t X1   = alloc(134217728ull);   // x1 f32 token-major
  const size_t HBUF = alloc(536870912ull);   // MLP hidden f32 NCHW

  // Weight + table setup
  f32_to_bf16<<<(786432 + 255) / 256, 256, 0, stream>>>(qkv_w, (unsigned short*)(ws + WQKV), 786432);
  f32_to_bf16<<<(262144 + 255) / 256, 256, 0, stream>>>(out_w, (unsigned short*)(ws + WOUT), 262144);
  f32_to_bf16<<<(1048576 + 255) / 256, 256, 0, stream>>>(pw1_w, (unsigned short*)(ws + WPW1), 1048576);
  f32_to_bf16<<<(1048576 + 255) / 256, 256, 0, stream>>>(pw2_w, (unsigned short*)(ws + WPW2), 1048576);
  setup_tbl<<<(3616 + 255) / 256, 256, 0, stream>>>(cpb_w1, cpb_w2, ls,
                                                    (float*)(ws + TBL), (float*)(ws + SCL));

  // LPU + window partition
  lpu_kernel<<<33554432 / 256, 256, 0, stream>>>(x, lpu_w, (float*)(ws + XL));
  win_gather<<<33554432 / 256, 256, 0, stream>>>((const float*)(ws + XL), (unsigned short*)(ws + XWIN));

  // QKV projection (65536x512 @ 512x1536) -> bf16
  gemm_bf16_wmma<<<dim3(1024, 6), 256, 0, stream>>>(
      (const unsigned short*)(ws + XWIN), (const unsigned short*)(ws + WQKV),
      (void*)(ws + QKV), 65536, 1536, 512, 2, 0);

  // Cosine-norm rearrange
  qkv_rearrange<<<65536, 512, 0, stream>>>((const unsigned short*)(ws + QKV),
                                           (unsigned short*)(ws + QN),
                                           (unsigned short*)(ws + KN),
                                           (unsigned short*)(ws + VT));

  // Attention
  attn_logits<<<dim3(256, 1024), 32, 0, stream>>>(
      (const unsigned short*)(ws + QN), (const unsigned short*)(ws + KN),
      (const float*)(ws + TBL), (const float*)(ws + SCL), (unsigned short*)(ws + SBUF));
  softmax_heads<<<16384, 256, 0, stream>>>((unsigned short*)(ws + SBUF));
  attn_pv<<<dim3(128, 1024), 32, 0, stream>>>(
      (const unsigned short*)(ws + SBUF), (const unsigned short*)(ws + VT), (float*)(ws + OBUF));

  // Output projection (65536x512 @ 512x512)
  f32_to_bf16<<<33554432 / 256, 256, 0, stream>>>((const float*)(ws + OBUF),
                                                  (unsigned short*)(ws + XWIN), 33554432);
  gemm_bf16_wmma<<<dim3(1024, 2), 256, 0, stream>>>(
      (const unsigned short*)(ws + XWIN), (const unsigned short*)(ws + WOUT),
      (void*)(ws + OBUF), 65536, 512, 512, 0, 0);

  // Residual 1 + LN (scrambled un-partition)
  ln1_res<<<65536, 256, 0, stream>>>((const float*)(ws + OBUF), (const float*)(ws + XL),
                                     gamma1, beta1, rs1,
                                     (float*)(ws + X1), (unsigned short*)(ws + XWIN));

  // MLP: pw1 (-> NCHW fp32), dwconv+StarReLU (-> bf16 tokens), pw2
  gemm_bf16_wmma<<<dim3(1024, 8), 256, 0, stream>>>(
      (const unsigned short*)(ws + XWIN), (const unsigned short*)(ws + WPW1),
      (void*)(ws + HBUF), 65536, 2048, 512, 1, 2048);
  mlp_dw_star<<<134217728 / 256, 256, 0, stream>>>((const float*)(ws + HBUF), dw_w, sr_s, sr_b,
                                                   (unsigned short*)(ws + QKV));
  gemm_bf16_wmma<<<dim3(1024, 2), 256, 0, stream>>>(
      (const unsigned short*)(ws + QKV), (const unsigned short*)(ws + WPW2),
      (void*)(ws + OBUF), 65536, 512, 2048, 0, 0);

  // Residual 2 + final LN -> NCHW output
  ln2_out<<<65536, 256, 0, stream>>>((const float*)(ws + OBUF), (const float*)(ws + X1),
                                     gamma2, beta2, rs2, (float*)d_out);
}
